// DynamicsModel_88072599372545
// MI455X (gfx1250) — matrix operand order repaired and checked by reading, not compiled, not run on hardware
//
#include <hip/hip_runtime.h>

// MAML inner-loop (5 SGD steps) + query loss for a 96->512->512->64 MLP.
// One workgroup (8 waves, 256 threads) per task; all matmuls via
// v_wmma_f32_16x16x32_bf16 (fp32 master params, bf16 operands, f32 accum).
//
// WMMA A-operand loads are contiguous b128 accesses: transposed copies
// (H^T, W^T) are produced in the GEMM epilogues, exploiting the fact that the
// WMMA D-layout is column-striped (lane = column, 8 consecutive rows), so a
// transposed tile store is one contiguous v8bf (16 B) store per lane.
//
// Workspace per task: fp32 params (1,380,608 B) + bf16 tensors (4,096,000 B)
//   = 5,476,608 B; x64 tasks + 256 B task-loss tail ~= 335 MiB.
//   Hot inner-loop tensors stay close to L2-resident (192 MB L2, 23.3 TB/s HBM).

#define METAB 64
#define NB    512
#define OBS   64
#define ACTD  32
#define IND   96
#define HIDD  512
#define STEPS 5
#define LRATE 0.01f

typedef __attribute__((ext_vector_type(16))) __bf16 v16bf;
typedef __attribute__((ext_vector_type(8)))  __bf16 v8bf;
typedef __attribute__((ext_vector_type(8)))  float  v8f;
typedef __attribute__((ext_vector_type(4)))  float  v4f;

#define PARAM_FLOATS 345152   // 49152 + 512 + 262144 + 512 + 32768 + 64
#define PARAM_BYTES  ((size_t)PARAM_FLOATS * 4)
#define XS_ELEMS ((size_t)NB * IND)      // 49152
#define H_ELEMS  ((size_t)NB * HIDD)     // 262144
#define DP_ELEMS ((size_t)NB * OBS)      // 32768
// bf16 region: Xs, XsT, Xq, H0, H0T, H1, H1T, dH1, dH0, dP, W1T, W2T
#define BF16_ELEMS (3*XS_ELEMS + 7*H_ELEMS + 2*DP_ELEMS)   // 2,048,000
#define TASK_STRIDE (PARAM_BYTES + BF16_ELEMS*2)           // 5,476,608 B

__device__ __forceinline__ void fb() {
  __threadfence();     // device-scope fence: flush stores to L2, invalidate read path
  __syncthreads();
}

// ---- WMMA fragment loaders (layouts per CDNA5 ISA 7.12.2) -------------------

// A (16x32 bf16): lane m (0-15) rows; lanes<16 hold K 0-7 & 16-23, lanes>=16 hold 8-15 & 24-31.
__device__ __forceinline__ v16bf ldA_bf16(const __bf16* src, int ld, int m0, int k0, int lane) {
  const int r  = m0 + (lane & 15);
  const int kb = k0 + ((lane & 16) ? 8 : 0);
  const __bf16* p = src + (size_t)r * ld + kb;
  v8bf lo = *(const v8bf*)(p);
  v8bf hi = *(const v8bf*)(p + 16);
  v16bf a;
#pragma unroll
  for (int e = 0; e < 8; ++e) { a[e] = lo[e]; a[e + 8] = hi[e]; }
  return a;
}

// B (32x16 bf16): lane n (0-15) column; lanes<16 hold K 0-7 & 16-23, lanes>=16 hold 8-15 & 24-31.
__device__ __forceinline__ v16bf ldB_bf16(const __bf16* src, int ld, int k0, int n0, int lane) {
  const __bf16* p = src + (size_t)(k0 + ((lane & 16) ? 8 : 0)) * ld + n0 + (lane & 15);
  v16bf b;
#pragma unroll
  for (int e = 0; e < 8; ++e) { b[e] = p[(size_t)e * ld]; b[e + 8] = p[(size_t)(e + 16) * ld]; }
  return b;
}

// B from fp32 row-major weights: lane column n, K in ISA order, convert to bf16.
__device__ __forceinline__ v16bf ldB_f32(const float* src, int ld, int k0, int n0, int lane) {
  const float* p = src + (size_t)(k0 + ((lane & 16) ? 8 : 0)) * ld + n0 + (lane & 15);
  v16bf b;
#pragma unroll
  for (int e = 0; e < 8; ++e) { b[e] = (__bf16)p[(size_t)e * ld]; b[e + 8] = (__bf16)p[(size_t)(e + 16) * ld]; }
  return b;
}

__device__ __forceinline__ v8f wmma_bf16(v16bf a, v16bf b, v8f c) {
  // (neg_a, A, neg_b, B, c_mod, C, reuse_a, reuse_b)
  return __builtin_amdgcn_wmma_f32_16x16x32_bf16(false, a, false, b, (short)0, c, false, false);
}

// ---- GEMM phases (8 waves stripe the 16x16 output tiles) --------------------

// H[NB x N] = relu(A[NB x K] * W[K x N] + b); dual-store H (row-major, b16
// scatter) and HT[N x NB] (one contiguous v8bf store per lane).
__device__ void fwd_relu(const __bf16* A, const float* W, const float* bias,
                         __bf16* H, __bf16* HT, int K, int N, int wave, int lane) {
  const int tilesN = N >> 4;
  const int total  = (NB >> 4) * tilesN;
  for (int t = wave; t < total; t += 8) {
    const int m0 = (t / tilesN) << 4;
    const int n0 = (t % tilesN) << 4;
    v8f acc = {};
    for (int k0 = 0; k0 < K; k0 += 32)
      acc = wmma_bf16(ldA_bf16(A, K, m0, k0, lane), ldB_f32(W, N, k0, n0, lane), acc);
    const int col = n0 + (lane & 15);
    const int rb  = m0 + ((lane & 16) ? 8 : 0);
    const float bv = bias[col];
    v8bf ht;
#pragma unroll
    for (int r = 0; r < 8; ++r) {
      float v = acc[r] + bv;
      const __bf16 h = (__bf16)(v > 0.f ? v : 0.f);
      H[(size_t)(rb + r) * N + col] = h;
      ht[r] = h;
    }
    *(v8bf*)(HT + (size_t)col * NB + rb) = ht;   // contiguous transposed store
  }
}

// dP = (2/(NB*OBS)) * (H1*W2 + b2 - delta), fused output-layer + loss-grad.
__device__ void out_grad(const __bf16* H1, const float* W2, const float* b2,
                         const float* delta, __bf16* dP, int wave, int lane) {
  const int tilesN = OBS >> 4;
  const int total  = (NB >> 4) * tilesN;
  const float gs = 2.0f / (float)(NB * OBS);
  for (int t = wave; t < total; t += 8) {
    const int m0 = (t / tilesN) << 4;
    const int n0 = (t % tilesN) << 4;
    v8f acc = {};
    for (int k0 = 0; k0 < HIDD; k0 += 32)
      acc = wmma_bf16(ldA_bf16(H1, HIDD, m0, k0, lane), ldB_f32(W2, OBS, k0, n0, lane), acc);
    const int col = n0 + (lane & 15);
    const int rb  = m0 + ((lane & 16) ? 8 : 0);
    const float bv = b2[col];
#pragma unroll
    for (int r = 0; r < 8; ++r) {
      const size_t idx = (size_t)(rb + r) * OBS + col;
      dP[idx] = (__bf16)(gs * (acc[r] + bv - delta[idx]));
    }
  }
}

// Query loss: per-thread sum of (pred - delta)^2 over assigned tiles.
__device__ float out_qloss(const __bf16* H1, const float* W2, const float* b2,
                           const float* delta, int wave, int lane) {
  const int tilesN = OBS >> 4;
  const int total  = (NB >> 4) * tilesN;
  float lsum = 0.f;
  for (int t = wave; t < total; t += 8) {
    const int m0 = (t / tilesN) << 4;
    const int n0 = (t % tilesN) << 4;
    v8f acc = {};
    for (int k0 = 0; k0 < HIDD; k0 += 32)
      acc = wmma_bf16(ldA_bf16(H1, HIDD, m0, k0, lane), ldB_f32(W2, OBS, k0, n0, lane), acc);
    const int col = n0 + (lane & 15);
    const int rb  = m0 + ((lane & 16) ? 8 : 0);
    const float bv = b2[col];
#pragma unroll
    for (int r = 0; r < 8; ++r) {
      const float d = acc[r] + bv - delta[(size_t)(rb + r) * OBS + col];
      lsum += d * d;
    }
  }
  return lsum;
}

// dOut[NB x N] = (dG[NB x Kg] * W^T) ∘ relu'(H); W^T is a pre-built bf16 copy
// WT[Kg x N] (contiguous B loads), mask read from HT (contiguous v8bf load).
__device__ void dact(const __bf16* dG, const __bf16* WT, const __bf16* HmT,
                     __bf16* dOut, int Kg, int N, int wave, int lane) {
  const int tilesN = N >> 4;
  const int total  = (NB >> 4) * tilesN;
  for (int t = wave; t < total; t += 8) {
    const int m0 = (t / tilesN) << 4;
    const int n0 = (t % tilesN) << 4;
    v8f acc = {};
    for (int k0 = 0; k0 < Kg; k0 += 32)
      acc = wmma_bf16(ldA_bf16(dG, Kg, m0, k0, lane), ldB_bf16(WT, N, k0, n0, lane), acc);
    const int col = n0 + (lane & 15);
    const int rb  = m0 + ((lane & 16) ? 8 : 0);
    v8bf hm = *(const v8bf*)(HmT + (size_t)col * NB + rb);
#pragma unroll
    for (int r = 0; r < 8; ++r) {
      const float v = ((float)hm[r] > 0.f) ? acc[r] : 0.f;
      dOut[(size_t)(rb + r) * N + col] = (__bf16)v;
    }
  }
}

// W[M x N] -= lr * Act^T * dG, K = NB; ActT[M x NB] bf16 (contiguous A loads),
// dG[NB x N] bf16 (contiguous B loads). Fused SGD update; optionally refresh
// the bf16 transposed copy WT[N x M] for next step's dact (contiguous store).
__device__ void wgrad_update(const __bf16* ActT, const __bf16* dG, float* W, __bf16* WT,
                             int M, int N, int wave, int lane) {
  const int tilesN = N >> 4;
  const int total  = (M >> 4) * tilesN;
  for (int t = wave; t < total; t += 8) {
    const int m0 = (t / tilesN) << 4;
    const int n0 = (t % tilesN) << 4;
    v8f acc = {};
    for (int k0 = 0; k0 < NB; k0 += 32)
      acc = wmma_bf16(ldA_bf16(ActT, NB, m0, k0, lane), ldB_bf16(dG, N, k0, n0, lane), acc);
    const int col = n0 + (lane & 15);
    const int rb  = m0 + ((lane & 16) ? 8 : 0);
    v8bf wt;
#pragma unroll
    for (int r = 0; r < 8; ++r) {
      const size_t idx = (size_t)(rb + r) * N + col;
      const float w = W[idx] - LRATE * acc[r];
      W[idx] = w;
      wt[r] = (__bf16)w;
    }
    if (WT) *(v8bf*)(WT + (size_t)col * M + rb) = wt;   // contiguous transposed store
  }
}

__device__ void bias_update(const __bf16* dG, float* b, int N, int tid) {
  for (int n = tid; n < N; n += 256) {
    float s = 0.f;
    for (int r = 0; r < NB; ++r) s += (float)dG[(size_t)r * N + n];
    b[n] -= LRATE * s;
  }
}

// ---- main per-task kernel ---------------------------------------------------

__global__ __launch_bounds__(256)
void maml_task_kernel(const float* past_obs, const float* past_act, const float* past_delta,
                      const float* fut_obs,  const float* fut_act,  const float* fut_delta,
                      const float* gW0, const float* gb0, const float* gW1, const float* gb1,
                      const float* gW2, const float* gb2,
                      float* task_losses, unsigned char* ws) {
  const int task = blockIdx.x;
  const int tid  = threadIdx.x;
  const int wave = tid >> 5;
  const int lane = tid & 31;
  __shared__ float sred[256];

  unsigned char* tb = ws + (size_t)task * TASK_STRIDE;
  float* W0 = (float*)tb;
  float* b0 = W0 + 49152;
  float* W1 = b0 + 512;
  float* b1 = W1 + 262144;
  float* W2 = b1 + 512;
  float* b2 = W2 + 32768;
  __bf16* Xs  = (__bf16*)(tb + PARAM_BYTES);
  __bf16* XsT = Xs  + XS_ELEMS;
  __bf16* Xq  = XsT + XS_ELEMS;
  __bf16* H0  = Xq  + XS_ELEMS;
  __bf16* H0T = H0  + H_ELEMS;
  __bf16* H1  = H0T + H_ELEMS;
  __bf16* H1T = H1  + H_ELEMS;
  __bf16* dH1 = H1T + H_ELEMS;
  __bf16* dH0 = dH1 + H_ELEMS;
  __bf16* dP  = dH0 + H_ELEMS;
  __bf16* W1T = dP  + DP_ELEMS;   // [HIDD x HIDD]
  __bf16* W2T = W1T + H_ELEMS;    // [OBS x HIDD]

  const float* po = past_obs   + (size_t)task * NB * OBS;
  const float* pa = past_act   + (size_t)task * NB * ACTD;
  const float* pd = past_delta + (size_t)task * NB * OBS;
  const float* fo = fut_obs    + (size_t)task * NB * OBS;
  const float* fa = fut_act    + (size_t)task * NB * ACTD;
  const float* fd = fut_delta  + (size_t)task * NB * OBS;

  // ---- setup: per-task fp32 param copy, bf16 transposed weight copies,
  //      bf16 concat inputs (row-major + transposed support inputs).
  for (int i = tid; i < 49152;  i += 256) W0[i] = gW0[i];
  for (int i = tid; i < 512;    i += 256) b0[i] = gb0[i];
  for (int i = tid; i < 262144; i += 256) W1[i] = gW1[i];
  for (int i = tid; i < 512;    i += 256) b1[i] = gb1[i];
  for (int i = tid; i < 32768;  i += 256) W2[i] = gW2[i];
  if (tid < 64) b2[tid] = gb2[tid];
  for (int i = tid; i < HIDD * HIDD; i += 256) {    // W1T[k][n] = W1[n][k]
    const int k = i / HIDD, n = i % HIDD;
    W1T[i] = (__bf16)gW1[(size_t)n * HIDD + k];
  }
  for (int i = tid; i < OBS * HIDD; i += 256) {     // W2T[k][n] = W2[n][k]
    const int k = i / HIDD, n = i % HIDD;
    W2T[i] = (__bf16)gW2[(size_t)n * OBS + k];
  }
  for (int i = tid; i < NB * OBS; i += 256) {
    const int r = i / OBS, c = i % OBS;
    const __bf16 vs = (__bf16)po[i];
    Xs[(size_t)r * IND + c]  = vs;
    XsT[(size_t)c * NB + r]  = vs;
    Xq[(size_t)r * IND + c]  = (__bf16)fo[i];
  }
  for (int i = tid; i < NB * ACTD; i += 256) {
    const int r = i / ACTD, c = i % ACTD;
    const __bf16 vs = (__bf16)pa[i];
    Xs[(size_t)r * IND + OBS + c]       = vs;
    XsT[(size_t)(OBS + c) * NB + r]     = vs;
    Xq[(size_t)r * IND + OBS + c]       = (__bf16)fa[i];
  }
  fb();

  // ---- 5 inner SGD steps on the support set
  for (int step = 0; step < STEPS; ++step) {
    fwd_relu(Xs, W0, b0, H0, H0T, IND,  HIDD, wave, lane); fb();
    fwd_relu(H0, W1, b1, H1, H1T, HIDD, HIDD, wave, lane); fb();
    out_grad(H1, W2, b2, pd, dP, wave, lane);              fb();
    dact(dP,  W2T, H1T, dH1, OBS,  HIDD, wave, lane);      fb();  // before W2 update
    dact(dH1, W1T, H0T, dH0, HIDD, HIDD, wave, lane);      fb();  // before W1 update
    wgrad_update(H1T, dP,  W2, W2T, HIDD, OBS,  wave, lane);      // W2 -= lr*H1^T dP
    wgrad_update(H0T, dH1, W1, W1T, HIDD, HIDD, wave, lane);      // W1 -= lr*H0^T dH1
    wgrad_update(XsT, dH0, W0, nullptr, IND, HIDD, wave, lane);   // W0 -= lr*Xs^T dH0
    bias_update(dP,  b2, OBS,  tid);
    bias_update(dH1, b1, HIDD, tid);
    bias_update(dH0, b0, HIDD, tid);
    fb();
  }

  // ---- query forward + MSE with adapted params
  fwd_relu(Xq, W0, b0, H0, H0T, IND,  HIDD, wave, lane); fb();
  fwd_relu(H0, W1, b1, H1, H1T, HIDD, HIDD, wave, lane); fb();
  float lsum = out_qloss(H1, W2, b2, fd, wave, lane);

  sred[tid] = lsum;
  __syncthreads();
  for (int s = 128; s > 0; s >>= 1) {
    if (tid < s) sred[tid] += sred[tid + s];
    __syncthreads();
  }
  if (tid == 0) task_losses[task] = sred[0] / (float)(NB * OBS);
}

// deterministic final mean over tasks
__global__ void maml_reduce_kernel(const float* tl, float* out) {
  if (threadIdx.x == 0 && blockIdx.x == 0) {
    float s = 0.f;
    for (int i = 0; i < METAB; ++i) s += tl[i];
    *out = s / (float)METAB;
  }
}

extern "C" void kernel_launch(void* const* d_in, const int* in_sizes, int n_in,
                              void* d_out, int out_size, void* d_ws, size_t ws_size,
                              hipStream_t stream) {
  (void)in_sizes; (void)n_in; (void)out_size; (void)ws_size;
  const float* past_obs   = (const float*)d_in[0];
  const float* past_act   = (const float*)d_in[1];
  const float* past_delta = (const float*)d_in[2];
  const float* fut_obs    = (const float*)d_in[3];
  const float* fut_act    = (const float*)d_in[4];
  const float* fut_delta  = (const float*)d_in[5];
  const float* W0 = (const float*)d_in[6];
  const float* b0 = (const float*)d_in[7];
  const float* W1 = (const float*)d_in[8];
  const float* b1 = (const float*)d_in[9];
  const float* W2 = (const float*)d_in[10];
  const float* b2 = (const float*)d_in[11];

  unsigned char* ws = (unsigned char*)d_ws;
  float* task_losses = (float*)(ws + (size_t)METAB * TASK_STRIDE);

  maml_task_kernel<<<METAB, 256, 0, stream>>>(
      past_obs, past_act, past_delta, fut_obs, fut_act, fut_delta,
      W0, b0, W1, b1, W2, b2, task_losses, ws);
  maml_reduce_kernel<<<1, 32, 0, stream>>>(task_losses, (float*)d_out);
}
